// NCF_Feature_38208029065467
// MI455X (gfx1250) — matrix operand order repaired and checked by reading, not compiled, not run on hardware
//
#include <hip/hip_runtime.h>

typedef __attribute__((ext_vector_type(16))) _Float16 v16h;
typedef __attribute__((ext_vector_type(8)))  float    v8f;
typedef __attribute__((ext_vector_type(4)))  float    fvec4;

#define B_ROWS  16384
#define NUM_TAG 4096
#define DIM_F   128
#define EPS     1e-5f

// ---------------- workspace layout (bytes) ----------------
// feat  : f32 [16384][32]   raw features (cols 0-9 z_user, 10-14 emb_u, 15-24 z_item, 25-29 emb_i)
// stats : f32 [64]          sum_u[16], sumsq_u[16], sum_i[16], sumsq_i[16]
// coef  : f32 [64]          a[32], c[32]   per-feature-column affine for batchnorm
// EiT   : f16 [16][4096]    (0.1*E_item_tag)^T zero-padded to 16 cols
// EuT   : f16 [16][4096]    E_user_tag^T zero-padded
// WuT   : f16 [16][128]     W_user zero-padded (already N-major: W[n][k])
// WiT   : f16 [16][128]
static constexpr size_t FEAT_OFF  = 0;
static constexpr size_t STATS_OFF = FEAT_OFF + (size_t)B_ROWS * 32 * 4;
static constexpr size_t COEF_OFF  = STATS_OFF + 64 * 4;
static constexpr size_t EIT_OFF   = COEF_OFF + 64 * 4;
static constexpr size_t EUT_OFF   = EIT_OFF + (size_t)16 * NUM_TAG * 2;
static constexpr size_t WUT_OFF   = EUT_OFF + (size_t)16 * NUM_TAG * 2;
static constexpr size_t WIT_OFF   = WUT_OFF + (size_t)16 * DIM_F * 2;

// ---------------- prep: f16 transposed/padded operand copies + zero stats ----------------
__global__ __launch_bounds__(256) void prep_kernel(
    const float* __restrict__ E_user_tag, const float* __restrict__ E_item_tag,
    const float* __restrict__ W_user, const float* __restrict__ W_item,
    _Float16* __restrict__ EuT, _Float16* __restrict__ EiT,
    _Float16* __restrict__ WuT, _Float16* __restrict__ WiT,
    float* __restrict__ stats)
{
    int idx = blockIdx.x * 256 + threadIdx.x;
    if (idx < 64) stats[idx] = 0.f;
    if (idx < 16 * NUM_TAG) {
        int n = idx >> 12;          // 0..15
        int k = idx & (NUM_TAG - 1);
        EuT[idx] = (n < 5) ? (_Float16)E_user_tag[k * 5 + n] : (_Float16)0.f;
        EiT[idx] = (n < 5) ? (_Float16)(0.1f * E_item_tag[k * 5 + n]) : (_Float16)0.f;
    }
    if (idx < 16 * DIM_F) {
        int n = idx >> 7;
        WuT[idx] = (n < 10) ? (_Float16)W_user[idx] : (_Float16)0.f;
        WiT[idx] = (n < 10) ? (_Float16)W_item[idx] : (_Float16)0.f;
    }
}

// A fragment (16x32, MxK) from row-major f32, rows r0..r0+15, cols k0..k0+31.
// ISA layout: lane m=L&15, h=L>>4; halves 0..7 -> K=8h+0..7, halves 8..15 -> K=16+8h+0..7.
__device__ __forceinline__ v16h load_a_f32(const float* __restrict__ X, int ld,
                                           int r0, int k0, int m, int h) {
    const float* p = X + (size_t)(r0 + m) * ld + k0 + 8 * h;
    fvec4 f0 = __builtin_nontemporal_load((const fvec4*)(p + 0));
    fvec4 f1 = __builtin_nontemporal_load((const fvec4*)(p + 4));
    fvec4 f2 = __builtin_nontemporal_load((const fvec4*)(p + 16));
    fvec4 f3 = __builtin_nontemporal_load((const fvec4*)(p + 20));
    v16h a;
    a[0]=(_Float16)f0[0]; a[1]=(_Float16)f0[1]; a[2]=(_Float16)f0[2]; a[3]=(_Float16)f0[3];
    a[4]=(_Float16)f1[0]; a[5]=(_Float16)f1[1]; a[6]=(_Float16)f1[2]; a[7]=(_Float16)f1[3];
    a[8]=(_Float16)f2[0]; a[9]=(_Float16)f2[1]; a[10]=(_Float16)f2[2]; a[11]=(_Float16)f2[3];
    a[12]=(_Float16)f3[0]; a[13]=(_Float16)f3[1]; a[14]=(_Float16)f3[2]; a[15]=(_Float16)f3[3];
    return a;
}

// B fragment (32x16, KxN) from N-major f16 [16][ldk]: lane n=L&15, h=L>>4; halves 0..7 -> K=8h+0..7, halves 8..15 -> K=16+8h+0..7.
__device__ __forceinline__ v16h load_b_f16(const _Float16* __restrict__ ET, int ldk,
                                           int k0, int n, int h) {
    typedef __attribute__((ext_vector_type(8))) _Float16 v8h_;
    const _Float16* p = ET + (size_t)n * ldk + k0 + 8 * h;
    v8h_ lo = *(const v8h_*)(p);
    v8h_ hi = *(const v8h_*)(p + 16);
    return __builtin_shufflevector(lo, hi, 0,1,2,3,4,5,6,7,8,9,10,11,12,13,14,15);
}

// ---------------- main streaming kernel: 4 waves/block, 16 rows per wave ----------------
__global__ __launch_bounds__(128) void fused_main(
    const float* __restrict__ user_feature, const float* __restrict__ item_feature,
    const float* __restrict__ user_tag, const float* __restrict__ item_tag,
    const float* __restrict__ b_user, const float* __restrict__ b_item,
    const _Float16* __restrict__ EiT, const _Float16* __restrict__ EuT,
    const _Float16* __restrict__ WuT, const _Float16* __restrict__ WiT,
    float* __restrict__ feat, float* __restrict__ stats)
{
    __shared__ float sstats[64];
    int t = threadIdx.x;
    if (t < 64) sstats[t] = 0.f;
    __syncthreads();

    int wave = t >> 5, lane = t & 31;
    int m = lane & 15, h = lane >> 4;
    int r0 = (blockIdx.x * 4 + wave) * 16;

    // ---- tag embedding GEMMs: [16,4096] x [4096,16pad], K-loop 128 WMMA steps each ----
    v8f acc_eu = {}; v8f acc_ei = {};
    for (int k0 = 0; k0 < NUM_TAG; k0 += 32) {
        v16h a0 = load_a_f32(user_tag, NUM_TAG, r0, k0, m, h);
        v16h b0 = load_b_f16(EiT, NUM_TAG, k0, m, h);
        acc_eu = __builtin_amdgcn_wmma_f32_16x16x32_f16(false, a0, false, b0,
                                                        (short)0, acc_eu, false, false);
        v16h a1 = load_a_f32(item_tag, NUM_TAG, r0, k0, m, h);
        v16h b1 = load_b_f16(EuT, NUM_TAG, k0, m, h);
        acc_ei = __builtin_amdgcn_wmma_f32_16x16x32_f16(false, a1, false, b1,
                                                        (short)0, acc_ei, false, false);
    }

    // ---- feature GEMMs: [16,128] x [128,16pad] ----
    v8f acc_zu = {}; v8f acc_zi = {};
    #pragma unroll
    for (int k0 = 0; k0 < DIM_F; k0 += 32) {
        v16h a0 = load_a_f32(user_feature, DIM_F, r0, k0, m, h);
        v16h b0 = load_b_f16(WuT, DIM_F, k0, m, h);
        acc_zu = __builtin_amdgcn_wmma_f32_16x16x32_f16(false, a0, false, b0,
                                                        (short)0, acc_zu, false, false);
        v16h a1 = load_a_f32(item_feature, DIM_F, r0, k0, m, h);
        v16h b1 = load_b_f16(WiT, DIM_F, k0, m, h);
        acc_zi = __builtin_amdgcn_wmma_f32_16x16x32_f16(false, a1, false, b1,
                                                        (short)0, acc_zi, false, false);
    }

    // C/D layout: VGPR v holds (row M = v + 8h, col N = lane&15)
    float bu = (m < 10) ? b_user[m] : 0.f;
    float bi = (m < 10) ? b_item[m] : 0.f;

    float su = 0.f, squ = 0.f, si = 0.f, sqi = 0.f;
    #pragma unroll
    for (int v = 0; v < 8; ++v) {
        float zu = acc_zu[v] + bu;
        float zi = acc_zi[v] + bi;
        acc_zu[v] = zu; acc_zi[v] = zi;
        su += zu; squ += zu * zu;
        si += zi; sqi += zi * zi;
    }
    if (m < 10) {
        atomicAdd(&sstats[m],      su);
        atomicAdd(&sstats[16 + m], squ);
        atomicAdd(&sstats[32 + m], si);
        atomicAdd(&sstats[48 + m], sqi);
    }

    #pragma unroll
    for (int v = 0; v < 8; ++v) {
        int r = r0 + v + 8 * h;
        float* fr = feat + (size_t)r * 32;
        if (m < 10) { fr[m] = acc_zu[v]; fr[15 + m] = acc_zi[v]; }
        if (m < 5)  { fr[10 + m] = acc_eu[v]; fr[25 + m] = acc_ei[v]; }
    }

    __syncthreads();
    if (t < 64) atomicAdd(&stats[t], sstats[t]);
}

// ---------------- batchnorm coefficients ----------------
__global__ void coef_kernel(const float* __restrict__ stats,
                            const float* __restrict__ g1, const float* __restrict__ be1,
                            const float* __restrict__ g2, const float* __restrict__ be2,
                            float* __restrict__ coef)
{
    int j = threadIdx.x;  // feature column 0..31
    float a, c;
    const float invB = 1.0f / (float)B_ROWS;
    if (j < 10) {
        float mu = stats[j] * invB;
        float var = stats[16 + j] * invB - mu * mu;
        float s = g1[j] * rsqrtf(var + EPS);
        a = s; c = be1[j] - s * mu;
    } else if (j < 15) {
        a = 1.f; c = 0.f;
    } else if (j < 25) {
        int q = j - 15;
        float mu = stats[32 + q] * invB;
        float var = stats[48 + q] * invB - mu * mu;
        float s = g2[q] * rsqrtf(var + EPS);
        a = s; c = be2[q] - s * mu;
    } else {
        a = (j < 30) ? 1.f : 0.f; c = 0.f;
    }
    coef[j] = a; coef[32 + j] = c;
}

// ---------------- per-row MLP: 30 -> 64 -> 32 -> 1 with ReLU ----------------
__global__ __launch_bounds__(256) void mlp_kernel(
    const float* __restrict__ feat, const float* __restrict__ coef,
    const float* __restrict__ W1, const float* __restrict__ b1,
    const float* __restrict__ W2, const float* __restrict__ b2,
    const float* __restrict__ W3, const float* __restrict__ b3,
    float* __restrict__ out)
{
    __shared__ float sW1[64 * 30], sW2[32 * 64];
    __shared__ float sb1[64], sb2[32], sW3[32], sCA[32], sCC[32], sb3;
    int t = threadIdx.x;
    for (int i = t; i < 64 * 30; i += 256) sW1[i] = W1[i];
    for (int i = t; i < 32 * 64; i += 256) sW2[i] = W2[i];
    if (t < 64) sb1[t] = b1[t];
    if (t < 32) { sb2[t] = b2[t]; sW3[t] = W3[t]; sCA[t] = coef[t]; sCC[t] = coef[32 + t]; }
    if (t == 0) sb3 = b3[0];
    __syncthreads();

    int r = blockIdx.x * 256 + t;
    const float* fr = feat + (size_t)r * 32;
    float f[30];
    #pragma unroll
    for (int j = 0; j < 30; ++j) f[j] = fmaf(sCA[j], fr[j], sCC[j]);

    float h1[64];
    #pragma unroll
    for (int o = 0; o < 64; ++o) {
        float s = sb1[o];
        #pragma unroll
        for (int k = 0; k < 30; ++k) s = fmaf(sW1[o * 30 + k], f[k], s);
        h1[o] = fmaxf(s, 0.f);
    }
    float h2[32];
    #pragma unroll
    for (int o = 0; o < 32; ++o) {
        float s = sb2[o];
        #pragma unroll
        for (int k = 0; k < 64; ++k) s = fmaf(sW2[o * 64 + k], h1[k], s);
        h2[o] = fmaxf(s, 0.f);
    }
    float s = sb3;
    #pragma unroll
    for (int k = 0; k < 32; ++k) s = fmaf(sW3[k], h2[k], s);
    out[r] = fmaxf(s, 0.f);
}

extern "C" void kernel_launch(void* const* d_in, const int* in_sizes, int n_in,
                              void* d_out, int out_size, void* d_ws, size_t ws_size,
                              hipStream_t stream) {
    const float* user_feature = (const float*)d_in[2];
    const float* item_feature = (const float*)d_in[3];
    const float* user_tag     = (const float*)d_in[4];
    const float* item_tag     = (const float*)d_in[5];
    const float* W_user       = (const float*)d_in[6];
    const float* b_user       = (const float*)d_in[7];
    const float* W_item       = (const float*)d_in[8];
    const float* b_item       = (const float*)d_in[9];
    const float* E_user_tag   = (const float*)d_in[10];
    const float* E_item_tag   = (const float*)d_in[11];
    const float* g1  = (const float*)d_in[12];
    const float* be1 = (const float*)d_in[13];
    const float* g2  = (const float*)d_in[14];
    const float* be2 = (const float*)d_in[15];
    const float* W1  = (const float*)d_in[16];
    const float* b1  = (const float*)d_in[17];
    const float* W2  = (const float*)d_in[18];
    const float* b2  = (const float*)d_in[19];
    const float* W3  = (const float*)d_in[20];
    const float* b3  = (const float*)d_in[21];

    char* ws = (char*)d_ws;
    float*     feat  = (float*)(ws + FEAT_OFF);
    float*     stats = (float*)(ws + STATS_OFF);
    float*     coef  = (float*)(ws + COEF_OFF);
    _Float16*  EiT   = (_Float16*)(ws + EIT_OFF);
    _Float16*  EuT   = (_Float16*)(ws + EUT_OFF);
    _Float16*  WuT   = (_Float16*)(ws + WUT_OFF);
    _Float16*  WiT   = (_Float16*)(ws + WIT_OFF);

    prep_kernel<<<(16 * NUM_TAG + 255) / 256, 256, 0, stream>>>(
        E_user_tag, E_item_tag, W_user, W_item, EuT, EiT, WuT, WiT, stats);

    fused_main<<<B_ROWS / 64, 128, 0, stream>>>(
        user_feature, item_feature, user_tag, item_tag,
        b_user, b_item, EiT, EuT, WuT, WiT, feat, stats);

    coef_kernel<<<1, 32, 0, stream>>>(stats, g1, be1, g2, be2, coef);

    mlp_kernel<<<B_ROWS / 256, 256, 0, stream>>>(
        feat, coef, W1, b1, W2, b2, W3, b3, (float*)d_out);
}
